// M_YOLO_69853348102837
// MI455X (gfx1250) — compile-verified, matrix operand order checked by reading, not run on hardware
//
#include <hip/hip_runtime.h>
#include <hip/hip_bf16.h>

typedef __attribute__((ext_vector_type(16))) _Float16 v16h;
typedef __attribute__((ext_vector_type(8)))  _Float16 v8h;
typedef __attribute__((ext_vector_type(8)))  float    v8f;
typedef __attribute__((ext_vector_type(4)))  int      v4i;

#define BB 8
#define SS 1024
#define DD 1024
#define HH 16
#define HD 64
#define QK_SCALE 0.125f   // 64^-0.5

#if __has_builtin(__builtin_amdgcn_global_load_async_to_lds_b128)
#define HAVE_ASYNC_COPY 1
#endif

__device__ __forceinline__ void wait_async0() {
#if __has_builtin(__builtin_amdgcn_s_wait_asynccnt)
    __builtin_amdgcn_s_wait_asynccnt(0);
#else
    asm volatile("s_wait_asynccnt 0" ::: "memory");
#endif
}

// ---------------------------------------------------------------------------
// WMMA wrapper: D = A(16x32 f16) * B(32x16 f16) + C(16x16 f32)
// ---------------------------------------------------------------------------
__device__ __forceinline__ v8f wmma_f16(v16h a, v16h b, v8f c) {
    return __builtin_amdgcn_wmma_f32_16x16x32_f16(
        false, a, false, b, (short)0, c, false, false);
}

// ---------------------------------------------------------------------------
// A/B fragment loader (16-bit operand, 16x32 tile), f16 source.
// Lane L: row/col = L&15, khalf = L>>4.
//   elems 0..7  -> K = khalf*8 + e ; elems 8..15 -> K = 16 + khalf*8 + e
// ---------------------------------------------------------------------------
__device__ __forceinline__ v16h frag_from_f16(const _Float16* __restrict__ p, int kh) {
    v16h f;
    const _Float16* a = p + kh * 8;
    const _Float16* b = p + 16 + kh * 8;
#pragma unroll
    for (int e = 0; e < 8; ++e) {
        f[e]     = a[e];
        f[e + 8] = b[e];
    }
    return f;
}

// ---------------------------------------------------------------------------
// Kernel 0: fp32 -> f16 conversion (8 elements / thread).
// ---------------------------------------------------------------------------
__global__ __launch_bounds__(256)
void cvt_f32_to_f16_kernel(const float* __restrict__ src,
                           _Float16* __restrict__ dst, int n8) {
    const int i = blockIdx.x * blockDim.x + threadIdx.x;
    if (i >= n8) return;
    const float* s = src + (size_t)i * 8;
    v8h h;
#pragma unroll
    for (int e = 0; e < 8; ++e) h[e] = (_Float16)s[e];
    *(v8h*)(dst + (size_t)i * 8) = h;
}

// ---------------------------------------------------------------------------
// Kernel 1: fused QKV projection, all-f16 operands.
//   Xh : [B*S, D] f16,  W*h : [D, D] f16 (out = X @ W^T + b)
//   Q/K/V out: f16 [B, H, S, HD]; Q scaled by QK_SCALE after bias.
// Block 256 = 8 waves; wave tile 16(M) x 64(N) (4 accumulators, A reuse x4).
// grid = (M/128, D/64, 3)
// ---------------------------------------------------------------------------
__global__ __launch_bounds__(256)
void qkv_proj_kernel(const _Float16* __restrict__ Xh,
                     const _Float16* __restrict__ Wqh, const float* __restrict__ bq,
                     const _Float16* __restrict__ Wkh, const float* __restrict__ bk,
                     const _Float16* __restrict__ Wvh, const float* __restrict__ bv,
                     _Float16* __restrict__ Qo, _Float16* __restrict__ Ko,
                     _Float16* __restrict__ Vo) {
    const int wave = threadIdx.x >> 5;
    const int lane = threadIdx.x & 31;
    const int m0 = blockIdx.x * 128 + wave * 16;
    const int n0 = blockIdx.y * 64;
    const int which = blockIdx.z;

    const _Float16* W;
    const float* bias;
    _Float16* out;
    float scale;
    if (which == 0)      { W = Wqh; bias = bq; out = Qo; scale = QK_SCALE; }
    else if (which == 1) { W = Wkh; bias = bk; out = Ko; scale = 1.0f; }
    else                 { W = Wvh; bias = bv; out = Vo; scale = 1.0f; }

    const int ln = lane & 15;
    const int kh = lane >> 4;
    const _Float16* arow = Xh + (size_t)(m0 + ln) * DD;
    const _Float16* wrow = W + (size_t)(n0 + ln) * DD;

    v8f acc[4] = {};
#pragma unroll 2
    for (int k0 = 0; k0 < DD; k0 += 32) {
        __builtin_prefetch(arow + k0 + 256, 0, 3);
        const v16h a = frag_from_f16(arow + k0, kh);
#pragma unroll
        for (int t = 0; t < 4; ++t) {
            const v16h b = frag_from_f16(wrow + (size_t)(16 * t) * DD + k0, kh);
            acc[t] = wmma_f16(a, b, acc[t]);
        }
    }

    // Epilogue: C layout -> lane holds col, rows r + 8*(lane>>4).
#pragma unroll
    for (int t = 0; t < 4; ++t) {
        const int n  = n0 + 16 * t + ln;
        const float bn = bias[n];
        const int h  = n >> 6;       // head
        const int hd = n & 63;       // dim within head
#pragma unroll
        for (int r = 0; r < 8; ++r) {
            const int m = m0 + r + 8 * kh;
            const int bidx = m >> 10;
            const int s    = m & (SS - 1);
            const float val = (acc[t][r] + bn) * scale;
            out[(((size_t)bidx * HH + h) * SS + s) * HD + hd] = (_Float16)val;
        }
    }
}

// ---------------------------------------------------------------------------
// Kernel 2: flash attention per (b, h).
//   Q/K/V: f16 [B, H, S, HD]; mask fp32 [B, 1, S, S] additive.
//   Output: f16 [B, S, D] row-major.
// Block 32 (one wave) handles a 16-row Q tile x full HD; kv in 32-wide steps.
// V tile staged to LDS with async global->LDS copies (overlaps score WMMAs).
// grid = (S/16, B*H)
// ---------------------------------------------------------------------------
__global__ __launch_bounds__(32)
void flash_attn_kernel(const _Float16* __restrict__ Q,
                       const _Float16* __restrict__ K,
                       const _Float16* __restrict__ V,
                       const float* __restrict__ mask,
                       _Float16* __restrict__ Ao) {
    __shared__ _Float16 pbuf[16 * 32];   // P tile staging (C-layout -> A-layout)
    __shared__ _Float16 vtile[32 * 64];  // V kv-tile (32 rows x HD)

    const int lane = threadIdx.x;
    const int ln = lane & 15;
    const int hi = lane >> 4;
    const int q0 = blockIdx.x * 16;
    const int bh = blockIdx.y;
    const int b  = bh >> 4;
    const int h  = bh & 15;

    const _Float16* Qb = Q + (size_t)bh * SS * HD;
    const _Float16* Kb = K + (size_t)bh * SS * HD;
    const _Float16* Vb = V + (size_t)bh * SS * HD;
    const float* maskb = mask + (size_t)b * SS * SS;

    const v16h qa0 = frag_from_f16(Qb + (size_t)(q0 + ln) * HD + 0, hi);
    const v16h qa1 = frag_from_f16(Qb + (size_t)(q0 + ln) * HD + 32, hi);

    float mrow[8], lrow[8];
    v8f o[4] = {};
#pragma unroll
    for (int r = 0; r < 8; ++r) { mrow[r] = -1e30f; lrow[r] = 0.0f; }

    for (int kv0 = 0; kv0 < SS; kv0 += 32) {
        // ---- kick off V tile copy to LDS (32x64 f16 = 4KB) ----
        const _Float16* vsrc = Vb + (size_t)kv0 * HD;
#if defined(HAVE_ASYNC_COPY)
#pragma unroll
        for (int i = 0; i < 8; ++i) {
            const int off = (i * 32 + lane) * 8;  // halves, 16B per lane
            __builtin_amdgcn_global_load_async_to_lds_b128(
                (v4i*)(vsrc + off), (v4i*)(vtile + off), 0, 0);
        }
#else
#pragma unroll
        for (int i = 0; i < 8; ++i) {
            const int off = (i * 32 + lane) * 8;
            *(v8h*)(vtile + off) = *(const v8h*)(vsrc + off);
        }
#endif

        // ---- scores (overlap with the async copy) ----
        v8f s0 = {}, s1 = {};
        {
            v16h kb0 = frag_from_f16(Kb + (size_t)(kv0 + ln) * HD + 0, hi);
            v16h kb1 = frag_from_f16(Kb + (size_t)(kv0 + ln) * HD + 32, hi);
            s0 = wmma_f16(qa0, kb0, s0);
            s0 = wmma_f16(qa1, kb1, s0);
            v16h kb2 = frag_from_f16(Kb + (size_t)(kv0 + 16 + ln) * HD + 0, hi);
            v16h kb3 = frag_from_f16(Kb + (size_t)(kv0 + 16 + ln) * HD + 32, hi);
            s1 = wmma_f16(qa0, kb2, s1);
            s1 = wmma_f16(qa1, kb3, s1);
        }

        // ---- additive mask + online softmax (rows live in 16-lane halves) ----
#pragma unroll
        for (int r = 0; r < 8; ++r) {
            const int m = q0 + r + 8 * hi;
            s0[r] += maskb[(size_t)m * SS + kv0 + ln];
            s1[r] += maskb[(size_t)m * SS + kv0 + 16 + ln];

            float mx = fmaxf(s0[r], s1[r]);
#pragma unroll
            for (int d = 1; d < 16; d <<= 1) mx = fmaxf(mx, __shfl_xor(mx, d, 32));
            const float mn    = fmaxf(mrow[r], mx);
            const float alpha = __expf(mrow[r] - mn);
            const float p0 = __expf(s0[r] - mn);
            const float p1 = __expf(s1[r] - mn);
            float rs = p0 + p1;
#pragma unroll
            for (int d = 1; d < 16; d <<= 1) rs += __shfl_xor(rs, d, 32);
            lrow[r] = lrow[r] * alpha + rs;
            mrow[r] = mn;
#pragma unroll
            for (int g = 0; g < 4; ++g) o[g][r] *= alpha;

            const int ml = r + 8 * hi;
            pbuf[ml * 32 + ln]      = (_Float16)p0;
            pbuf[ml * 32 + 16 + ln] = (_Float16)p1;
        }
        __syncthreads();
        const v16h pa = frag_from_f16(pbuf + ln * 32, hi);

        // ---- drain async copy, then PV from LDS ----
#if defined(HAVE_ASYNC_COPY)
        wait_async0();
#endif
        __syncthreads();

#pragma unroll
        for (int g = 0; g < 4; ++g) {
            v16h vb;
#pragma unroll
            for (int e = 0; e < 8; ++e) {
                vb[e]     = vtile[(hi * 8 + e) * HD      + g * 16 + ln];
                vb[e + 8] = vtile[(16 + hi * 8 + e) * HD + g * 16 + ln];
            }
            o[g] = wmma_f16(pa, vb, o[g]);
        }
        __syncthreads();  // vtile/pbuf reused next iteration
    }

    // ---- finalize: O /= l, write f16 [B, S, D] ----
#pragma unroll
    for (int r = 0; r < 8; ++r) {
        const float inv = 1.0f / lrow[r];
        const int m = q0 + r + 8 * hi;
        _Float16* dst = Ao + ((size_t)b * SS + m) * DD + h * HD;
#pragma unroll
        for (int g = 0; g < 4; ++g)
            dst[g * 16 + ln] = (_Float16)(o[g][r] * inv);
    }
}

// ---------------------------------------------------------------------------
// Kernel 3: output projection, all-f16 operands, fp32 result.
//   A : f16 [B*S, D], Woh : f16 [D, D], out = A @ Wo^T + bo (fp32)
// Block 256 = 8 waves; wave tile 16 x 64 (4 accumulators).
// grid = (M/128, D/64)
// ---------------------------------------------------------------------------
__global__ __launch_bounds__(256)
void out_proj_kernel(const _Float16* __restrict__ A,
                     const _Float16* __restrict__ Woh,
                     const float* __restrict__ bo,
                     float* __restrict__ out) {
    const int wave = threadIdx.x >> 5;
    const int lane = threadIdx.x & 31;
    const int m0 = blockIdx.x * 128 + wave * 16;
    const int n0 = blockIdx.y * 64;
    const int ln = lane & 15;
    const int kh = lane >> 4;

    const _Float16* arow = A   + (size_t)(m0 + ln) * DD;
    const _Float16* wrow = Woh + (size_t)(n0 + ln) * DD;

    v8f acc[4] = {};
#pragma unroll 2
    for (int k0 = 0; k0 < DD; k0 += 32) {
        __builtin_prefetch(arow + k0 + 256, 0, 3);
        const v16h a = frag_from_f16(arow + k0, kh);
#pragma unroll
        for (int t = 0; t < 4; ++t) {
            const v16h b = frag_from_f16(wrow + (size_t)(16 * t) * DD + k0, kh);
            acc[t] = wmma_f16(a, b, acc[t]);
        }
    }

#pragma unroll
    for (int t = 0; t < 4; ++t) {
        const int n = n0 + 16 * t + ln;
        const float bn = bo[n];
#pragma unroll
        for (int r = 0; r < 8; ++r) {
            const int m = m0 + r + 8 * kh;
            out[(size_t)m * DD + n] = acc[t][r] + bn;
        }
    }
}

// ---------------------------------------------------------------------------
// Host side
// ---------------------------------------------------------------------------
extern "C" void kernel_launch(void* const* d_in, const int* in_sizes, int n_in,
                              void* d_out, int out_size, void* d_ws, size_t ws_size,
                              hipStream_t stream) {
    const float* X    = (const float*)d_in[0];  // [B, S, D]
    const float* mask = (const float*)d_in[1];  // [B, 1, S, S]
    const float* Wq   = (const float*)d_in[2];
    const float* bq   = (const float*)d_in[3];
    const float* Wk   = (const float*)d_in[4];
    const float* bk   = (const float*)d_in[5];
    const float* Wv   = (const float*)d_in[6];
    const float* bv   = (const float*)d_in[7];
    const float* Wo   = (const float*)d_in[8];
    const float* bo   = (const float*)d_in[9];
    float* out = (float*)d_out;

    // Workspace layout (halves):
    //   Xh   : B*S*D        = 8388608
    //   Wqh/Wkh/Wvh/Woh : D*D = 1048576 each
    //   Qw/Kw/Vw : B*H*S*HD = 8388608 each
    //   Aw   : B*S*D        = 8388608
    const size_t nX = (size_t)BB * SS * DD;   // 8388608
    const size_t nW = (size_t)DD * DD;        // 1048576
    _Float16* Xh  = (_Float16*)d_ws;
    _Float16* Wqh = Xh + nX;
    _Float16* Wkh = Wqh + nW;
    _Float16* Wvh = Wkh + nW;
    _Float16* Woh = Wvh + nW;
    _Float16* Qw  = Woh + nW;
    _Float16* Kw  = Qw + nX;
    _Float16* Vw  = Kw + nX;
    _Float16* Aw  = Vw + nX;

    const int M = BB * SS;  // 8192

    // ---- convert fp32 activations + weights to f16 once ----
    cvt_f32_to_f16_kernel<<<(int)(nX / 8 / 256), 256, 0, stream>>>(X, Xh, (int)(nX / 8));
    cvt_f32_to_f16_kernel<<<(int)(nW / 8 / 256), 256, 0, stream>>>(Wq, Wqh, (int)(nW / 8));
    cvt_f32_to_f16_kernel<<<(int)(nW / 8 / 256), 256, 0, stream>>>(Wk, Wkh, (int)(nW / 8));
    cvt_f32_to_f16_kernel<<<(int)(nW / 8 / 256), 256, 0, stream>>>(Wv, Wvh, (int)(nW / 8));
    cvt_f32_to_f16_kernel<<<(int)(nW / 8 / 256), 256, 0, stream>>>(Wo, Woh, (int)(nW / 8));

    dim3 gqkv(M / 128, DD / 64, 3);
    qkv_proj_kernel<<<gqkv, 256, 0, stream>>>(Xh, Wqh, bq, Wkh, bk, Wvh, bv, Qw, Kw, Vw);

    dim3 gattn(SS / 16, BB * HH);
    flash_attn_kernel<<<gattn, 32, 0, stream>>>(Qw, Kw, Vw, mask, Aw);

    dim3 gout(M / 128, DD / 64);
    out_proj_kernel<<<gout, 256, 0, stream>>>(Aw, Woh, bo, out);
}
